// PointerGenerator_53575422050443
// MI455X (gfx1250) — compile-verified, matrix operand order checked by reading, not run on hardware
//
#include <hip/hip_runtime.h>
#include <hip/hip_bf16.h>
#include <math.h>

// Problem constants (match reference setup_inputs)
#define Hc 256
#define Ec 128
#define Vc 50257
#define Lc 400
#define Bc 32
#define MAXLEN 64
#define NPAD 50304   // 393 * 128, multiple of 128 for guard-free vocab GEMM
#define EPSC 1e-8f

typedef __attribute__((ext_vector_type(16))) __bf16 v16bf;
typedef __attribute__((ext_vector_type(8)))  float  v8f;
typedef __attribute__((ext_vector_type(4)))  unsigned int v4u;
typedef __attribute__((ext_vector_type(8)))  int  v8i;
typedef __attribute__((ext_vector_type(4)))  int  v4i;

__device__ __forceinline__ float sigmf(float x) { return 1.0f / (1.0f + __expf(-x)); }

__device__ __forceinline__ v8f wmma_bf16(v16bf a, v16bf b, v8f c) {
  // D = A(16x32) * B(32x16) + C, f32 accumulate
  return __builtin_amdgcn_wmma_f32_16x16x32_bf16(false, a, false, b, (short)0, c,
                                                 false, false);
}

// Fragment loader for A (row-major [M,K]) and B (weights stored [N,K] row-major,
// computing X @ W^T). wave32 layout: lane&15 selects row/col, lane>>4 selects K-half,
// 16 contiguous bf16 per lane.
__device__ __forceinline__ v16bf load_frag(const __bf16* base, int row0, int lda, int k0) {
  const int lane = threadIdx.x & 31;
  const __bf16* p = base + (size_t)(row0 + (lane & 15)) * lda + k0 + ((lane >> 4) << 4);
  return *(const v16bf*)p;
}

__device__ __forceinline__ unsigned int hash3(unsigned a, unsigned b, unsigned c) {
  unsigned h = a * 0x9E3779B1u ^ b * 0x85EBCA77u ^ c * 0xC2B2AE3Du;
  h ^= h >> 16; h *= 0x7FEB352Du; h ^= h >> 15; h *= 0x846CA68Bu; h ^= h >> 16;
  return h;
}

// ---------------------------------------------------------------------------
// Tensor Data Mover: async 2D tile load (bf16) global -> LDS.
// D# group0: count=1 | lds_addr | global_addr(57b) | type=2.
// D# group1: data_size=1 (2B), tensor dims large (no OOB), tile = tile_d0 x tile_d1,
//            dim0 stride in elements. Issued by one wave; tracked on TENSORcnt.
// ---------------------------------------------------------------------------
__device__ __forceinline__ void tdm_load_tile_bf16(unsigned lds_off, const __bf16* gptr,
                                                   int tile_d0 /*K elems*/,
                                                   int tile_d1 /*rows*/,
                                                   int d0_stride /*elems*/) {
  unsigned long long ga = (unsigned long long)(uintptr_t)gptr;
  unsigned ga_lo = (unsigned)__builtin_amdgcn_readfirstlane((int)(ga & 0xffffffffull));
  unsigned ga_hi = (unsigned)__builtin_amdgcn_readfirstlane((int)((ga >> 32) & 0x1ffffffull));
  unsigned lo    = (unsigned)__builtin_amdgcn_readfirstlane((int)lds_off);
  int td0 = __builtin_amdgcn_readfirstlane(tile_d0);
  int td1 = __builtin_amdgcn_readfirstlane(tile_d1);
  int st0 = __builtin_amdgcn_readfirstlane(d0_stride);

  v4u g0 = {1u,                      // count=1 (valid descriptor)
            lo,                      // lds_addr
            ga_lo,                   // global_addr[31:0]
            ga_hi | 0x80000000u};    // global_addr[56:32] | type=2<<30

  v8i g1;
  g1[0] = 0x00010000;                          // workgroup_mask=0, data_size=1 (2B)
  g1[1] = 0;                                   // tensor_dim0[63:48]=0 (dim0 = 1<<20)
  g1[2] = 0x10;                                // tensor_dim0[79:64]=0x10, tensor_dim1[95:80]=0
  g1[3] = 0x10 | (td0 << 16);                  // tensor_dim1[111:96]=0x10 | tile_dim0
  g1[4] = td1;                                 // tile_dim1, tile_dim2=0
  g1[5] = st0;                                 // tensor_dim0_stride[191:160]
  g1[6] = 0;                                   // stride0 hi, dim1_stride lo
  g1[7] = 0;                                   // dim1_stride hi
  v4i gz = {0, 0, 0, 0};
#if defined(__clang_major__) && (__clang_major__ >= 23)
  v8i gz8 = {0, 0, 0, 0, 0, 0, 0, 0};
  __builtin_amdgcn_tensor_load_to_lds(g0, g1, gz, gz, gz8, 0);
#else
  __builtin_amdgcn_tensor_load_to_lds(g0, g1, gz, gz, 0);
#endif
}

// ---------------------------------------------------------------------------
// Prep kernels
// ---------------------------------------------------------------------------
__global__ void convert_bf16_kernel(const float* __restrict__ s, __bf16* __restrict__ d, int n) {
  int idx = blockIdx.x * blockDim.x + threadIdx.x;
  if (idx < n) d[idx] = (__bf16)s[idx];
}

__global__ void convert_proW_kernel(const float* __restrict__ s, __bf16* __restrict__ d) {
  const int n = NPAD * 1024;
  for (int idx = blockIdx.x * blockDim.x + threadIdx.x; idx < n;
       idx += gridDim.x * blockDim.x) {
    int row = idx >> 10;
    d[idx] = (row < Vc) ? (__bf16)s[idx] : (__bf16)0.0f;
  }
}

__global__ void embed_src_kernel(const int* __restrict__ src, const float* __restrict__ emb_W,
                                 __bf16* __restrict__ xs) {
  int idx = blockIdx.x * blockDim.x + threadIdx.x;
  if (idx >= Lc * Bc * Ec) return;
  int l = idx / (Bc * Ec);
  int rem = idx - l * (Bc * Ec);
  int b = rem >> 7, e = rem & 127;
  xs[idx] = (__bf16)emb_W[(size_t)src[b * Lc + l] * Ec + e];
}

__global__ void dec_init_kernel(const int* __restrict__ start_symbol, int* __restrict__ nxt,
                                float* __restrict__ cov, float* __restrict__ covloss) {
  int idx = blockIdx.x * blockDim.x + threadIdx.x;
  if (idx < Bc) nxt[idx] = start_symbol[0];
  if (idx == 0) covloss[0] = 0.0f;
  if (idx < Bc * Lc) cov[idx] = 0.0f;
}

__global__ void hbf_init_kernel(const float* __restrict__ h, __bf16* __restrict__ h_bf) {
  int idx = blockIdx.x * blockDim.x + threadIdx.x;
  if (idx < Bc * 512) h_bf[idx] = (__bf16)h[idx];
}

__global__ void zero_kernel(float* __restrict__ p, int n) {
  for (int idx = blockIdx.x * blockDim.x + threadIdx.x; idx < n;
       idx += gridDim.x * blockDim.x) p[idx] = 0.0f;
}

__global__ void copy1_kernel(const float* __restrict__ s, float* __restrict__ d) {
  if (threadIdx.x == 0 && blockIdx.x == 0) d[0] = s[0];
}

// ---------------------------------------------------------------------------
// Persistent bidirectional LSTM encoder: 2 blocks (fwd/bwd) x 1024 threads.
// Per step: gates[32,1024] = [x|h] @ [Wih|Whh]^T + b via WMMA bf16, software-
// pipelined K-loop (fragments for step k+1 fetched before WMMAs of step k).
// Each wave owns one M-tile and one j-tile across all 4 gate chunks, so the
// cell nonlinearity runs in registers; c stays in VGPRs, h in LDS (bf16).
// ---------------------------------------------------------------------------
__global__ __launch_bounds__(1024)
void encoder_kernel(const __bf16* __restrict__ xs,   // [L, B, E]
                    const __bf16* WihF, const __bf16* WhhF,
                    const float* bihF, const float* bhhF,
                    const __bf16* WihB, const __bf16* WhhB,
                    const float* bihB, const float* bhhB,
                    float* __restrict__ memory,      // [B, L, 512]
                    float* __restrict__ dec_h, float* __restrict__ dec_c) {
  const int dir = blockIdx.x;
  const __bf16* Wih = dir ? WihB : WihF;
  const __bf16* Whh = dir ? WhhB : WhhF;
  const float*  bih = dir ? bihB : bihF;
  const float*  bhh = dir ? bhhB : bhhF;

  __shared__ __bf16 hs[32 * 264];   // h in bf16, padded stride

  const int tid   = threadIdx.x;
  const int lane  = tid & 31;
  const int w     = tid >> 5;         // 0..31 waves
  const int m0    = (w & 1) * 16;     // M-tile base (batch rows)
  const int tj    = w >> 1;           // 0..15: j-tile within gate chunk
  const int nlane = lane & 15;
  const int mb    = m0 + ((lane >> 4) << 3);  // this lane's C-frag row base
  const int j     = tj * 16 + nlane;          // hidden index 0..255

  for (int i = tid; i < 32 * 264; i += 1024) hs[i] = (__bf16)0.0f;
  __syncthreads();

  float creg[8];
#pragma unroll
  for (int r = 0; r < 8; ++r) creg[r] = 0.0f;

  const float bi_i = bih[j]       + bhh[j];
  const float bi_f = bih[256 + j] + bhh[256 + j];
  const float bi_g = bih[512 + j] + bhh[512 + j];
  const float bi_o = bih[768 + j] + bhh[768 + j];

  const v8f zero8 = {0.f, 0.f, 0.f, 0.f, 0.f, 0.f, 0.f, 0.f};

  for (int t = 0; t < Lc; ++t) {
    const int lt = dir ? (Lc - 1 - t) : t;
    const __bf16* xrow = xs + (size_t)lt * (Bc * Ec);
    v8f acc[4];
#pragma unroll
    for (int g = 0; g < 4; ++g) acc[g] = zero8;

    // 12 logical K-steps: 0..3 over x/Wih (K=128), 4..11 over h/Whh (K=256).
    v16bf aF[2];
    v16bf bF[2][4];
    aF[0] = load_frag(xrow, m0, Ec, 0);
#pragma unroll
    for (int g = 0; g < 4; ++g) bF[0][g] = load_frag(Wih, tj * 16 + g * 256, Ec, 0);

#pragma unroll
    for (int s = 0; s < 12; ++s) {
      const int cb = s & 1, nb = cb ^ 1;
      if (s + 1 < 12) {
        const int sn = s + 1;
        if (sn < 4) {
          aF[nb] = load_frag(xrow, m0, Ec, sn * 32);
#pragma unroll
          for (int g = 0; g < 4; ++g)
            bF[nb][g] = load_frag(Wih, tj * 16 + g * 256, Ec, sn * 32);
        } else {
          aF[nb] = load_frag(hs, m0, 264, (sn - 4) * 32);
#pragma unroll
          for (int g = 0; g < 4; ++g)
            bF[nb][g] = load_frag(Whh, tj * 16 + g * 256, Hc, (sn - 4) * 32);
        }
      }
#pragma unroll
      for (int g = 0; g < 4; ++g) acc[g] = wmma_bf16(aF[cb], bF[cb][g], acc[g]);
    }
    __syncthreads();   // all hs reads done before overwrite

#pragma unroll
    for (int r = 0; r < 8; ++r) {
      float iv = acc[0][r] + bi_i;
      float fv = acc[1][r] + bi_f;
      float gv = acc[2][r] + bi_g;
      float ov = acc[3][r] + bi_o;
      float cn = sigmf(fv) * creg[r] + sigmf(iv) * tanhf(gv);
      float hn = sigmf(ov) * tanhf(cn);
      creg[r] = cn;
      int b = mb + r;
      hs[b * 264 + j] = (__bf16)hn;
      memory[((size_t)b * Lc + lt) * 512 + dir * 256 + j] = hn;
      if (t == Lc - 1) {
        dec_h[b * 512 + dir * 256 + j] = hn;
        dec_c[b * 512 + dir * 256 + j] = cn;
      }
    }
    __syncthreads();
  }
}

// ---------------------------------------------------------------------------
// Decoder kernels (launched 64x)
// ---------------------------------------------------------------------------
__global__ void dec_embed_kernel(const float* __restrict__ emb_W, const int* __restrict__ nxt,
                                 float* __restrict__ emb_f, __bf16* __restrict__ emb_bf) {
  int idx = blockIdx.x * blockDim.x + threadIdx.x;
  if (idx >= Bc * Ec) return;
  int b = idx >> 7, e = idx & 127;
  float v = emb_W[(size_t)nxt[b] * Ec + e];
  emb_f[idx] = v;
  emb_bf[idx] = (__bf16)v;
}

// gates[32,2048] = [emb|h] @ [Wih|Whh]^T + bih + bhh (software-pipelined)
__global__ __launch_bounds__(256)
void dec_gates_kernel(const __bf16* __restrict__ emb_bf, const __bf16* __restrict__ h_bf,
                      const __bf16* __restrict__ Wih, const __bf16* __restrict__ Whh,
                      const float* __restrict__ bih, const float* __restrict__ bhh,
                      float* __restrict__ gates) {
  const int lane = threadIdx.x & 31;
  const int w = threadIdx.x >> 5;
  const int tile = blockIdx.x * 8 + w;    // 256 tiles = 2 M x 128 N
  const int m0 = (tile & 1) * 16;
  const int n0 = (tile >> 1) * 16;
  v8f acc = {0.f, 0.f, 0.f, 0.f, 0.f, 0.f, 0.f, 0.f};

  // 20 logical K-steps: 0..3 over emb/Wih (K=128), 4..19 over h/Whh (K=512).
  v16bf aP[2], bP[2];
  aP[0] = load_frag(emb_bf, m0, 128, 0);
  bP[0] = load_frag(Wih, n0, 128, 0);
#pragma unroll
  for (int s = 0; s < 20; ++s) {
    const int cb = s & 1, nb = cb ^ 1;
    if (s + 1 < 20) {
      const int sn = s + 1;
      if (sn < 4) {
        aP[nb] = load_frag(emb_bf, m0, 128, sn * 32);
        bP[nb] = load_frag(Wih, n0, 128, sn * 32);
      } else {
        aP[nb] = load_frag(h_bf, m0, 512, (sn - 4) * 32);
        bP[nb] = load_frag(Whh, n0, 512, (sn - 4) * 32);
      }
    }
    acc = wmma_bf16(aP[cb], bP[cb], acc);
  }
  const int col = n0 + (lane & 15);
  const int mbb = m0 + ((lane >> 4) << 3);
  const float bias = bih[col] + bhh[col];
#pragma unroll
  for (int r = 0; r < 8; ++r) gates[(mbb + r) * 2048 + col] = acc[r] + bias;
}

__global__ void dec_cell_kernel(const float* __restrict__ gates, float* __restrict__ h,
                                float* __restrict__ c, __bf16* __restrict__ h_bf,
                                __bf16* __restrict__ feature_bf) {
  int idx = blockIdx.x * blockDim.x + threadIdx.x;
  if (idx >= Bc * 512) return;
  int b = idx >> 9, jc = idx & 511;
  float iv = gates[b * 2048 + jc];
  float fv = gates[b * 2048 + 512 + jc];
  float gv = gates[b * 2048 + 1024 + jc];
  float ov = gates[b * 2048 + 1536 + jc];
  float cn = sigmf(fv) * c[idx] + sigmf(iv) * tanhf(gv);
  float hn = sigmf(ov) * tanhf(cn);
  c[idx] = cn;
  h[idx] = hn;
  h_bf[idx] = (__bf16)hn;
  feature_bf[b * 1024 + jc] = (__bf16)hn;   // feature[:, 0:512] = out
}

// Per-batch-row: scores, coverage softmax + covloss + cov update, plain softmax,
// context, pgen. One block of 256 threads per row.
__global__ __launch_bounds__(256)
void attention_kernel(const float* __restrict__ h, const float* __restrict__ memory,
                      float* __restrict__ cov, const float* __restrict__ cov_w,
                      const float* __restrict__ pgen_W, const float* __restrict__ pgen_b,
                      const float* __restrict__ emb_f,
                      float* __restrict__ att, float* __restrict__ ctx,
                      __bf16* __restrict__ feature_bf, float* __restrict__ pgen,
                      float* __restrict__ covloss) {
  __shared__ float outS[512], ctxS[512], sc[Lc], tmp[Lc], red[256];
  const int b = blockIdx.x, tid = threadIdx.x;

  for (int d = tid; d < 512; d += 256) outS[d] = h[b * 512 + d];
  __syncthreads();

  for (int l = tid; l < Lc; l += 256) {
    const float* mrow = memory + ((size_t)b * Lc + l) * 512;
    float s = 0.f;
    for (int d = 0; d < 512; ++d) s += outS[d] * mrow[d];
    sc[l] = s;
  }
  __syncthreads();

  const float cw = cov_w[0];
  // --- coverage softmax ---
  float lm = -1e30f;
  for (int l = tid; l < Lc; l += 256) lm = fmaxf(lm, sc[l] + cov[b * Lc + l] * cw);
  red[tid] = lm; __syncthreads();
  for (int s = 128; s > 0; s >>= 1) { if (tid < s) red[tid] = fmaxf(red[tid], red[tid + s]); __syncthreads(); }
  const float M1 = red[0]; __syncthreads();
  float ls = 0.f;
  for (int l = tid; l < Lc; l += 256) { float e = __expf(sc[l] + cov[b * Lc + l] * cw - M1); tmp[l] = e; ls += e; }
  red[tid] = ls; __syncthreads();
  for (int s = 128; s > 0; s >>= 1) { if (tid < s) red[tid] += red[tid + s]; __syncthreads(); }
  const float S1 = red[0]; __syncthreads();
  float clp = 0.f;
  for (int l = tid; l < Lc; l += 256) {
    float ac = tmp[l] / S1;
    float cv = cov[b * Lc + l];
    clp += fminf(ac, cv);
    cov[b * Lc + l] = cv + ac;
  }
  red[tid] = clp; __syncthreads();
  for (int s = 128; s > 0; s >>= 1) { if (tid < s) red[tid] += red[tid + s]; __syncthreads(); }
  if (tid == 0) atomicAdd(covloss, red[0] * (1.0f / Bc));
  __syncthreads();

  // --- plain softmax ---
  lm = -1e30f;
  for (int l = tid; l < Lc; l += 256) lm = fmaxf(lm, sc[l]);
  red[tid] = lm; __syncthreads();
  for (int s = 128; s > 0; s >>= 1) { if (tid < s) red[tid] = fmaxf(red[tid], red[tid + s]); __syncthreads(); }
  const float M2 = red[0]; __syncthreads();
  ls = 0.f;
  for (int l = tid; l < Lc; l += 256) { float e = __expf(sc[l] - M2); tmp[l] = e; ls += e; }
  red[tid] = ls; __syncthreads();
  for (int s = 128; s > 0; s >>= 1) { if (tid < s) red[tid] += red[tid + s]; __syncthreads(); }
  const float S2 = red[0]; __syncthreads();
  for (int l = tid; l < Lc; l += 256) { tmp[l] /= S2; att[b * Lc + l] = tmp[l]; }
  __syncthreads();

  // --- context ---
  for (int d = tid; d < 512; d += 256) {
    float a = 0.f;
    for (int l = 0; l < Lc; ++l) a += tmp[l] * memory[((size_t)b * Lc + l) * 512 + d];
    ctxS[d] = a;
    ctx[b * 512 + d] = a;
    feature_bf[b * 1024 + 512 + d] = (__bf16)a;  // feature[:, 512:1024] = context
  }
  __syncthreads();

  // --- pgen = sigmoid([context, out, ans_emb] . pgen_W + pgen_b) ---
  float ps = 0.f;
  for (int k = tid; k < 1152; k += 256) {
    float x = (k < 512) ? ctxS[k] : ((k < 1024) ? outS[k - 512] : emb_f[b * 128 + (k - 1024)]);
    ps += x * pgen_W[k];
  }
  red[tid] = ps; __syncthreads();
  for (int s = 128; s > 0; s >>= 1) { if (tid < s) red[tid] += red[tid + s]; __syncthreads(); }
  if (tid == 0) pgen[b] = sigmf(red[0] + pgen_b[0]);
}

__global__ void pointer_scatter_kernel(const int* __restrict__ src, const float* __restrict__ att,
                                       float* __restrict__ pointer) {
  int idx = blockIdx.x * blockDim.x + threadIdx.x;
  if (idx >= Bc * Lc) return;
  int b = idx / Lc, l = idx - b * Lc;
  atomicAdd(&pointer[(size_t)b * Vc + src[b * Lc + l]], att[idx]);
}

// ---------------------------------------------------------------------------
// logits[32, NPAD] = feature @ pro_W^T + pro_b  (bandwidth-dominant GEMM).
// TDM streams [128 x 64] bf16 weight tiles into double-buffered LDS (wave 0
// issues tensor_load_to_lds; S_WAIT_TENSORCNT + barrier to release consumers);
// 8 waves consume B-fragments from LDS, A stays in registers (L2-hot).
// ---------------------------------------------------------------------------
#define VK 64                      // K-chunk per TDM tile
#define VNC (1024 / VK)            // 16 chunks
__global__ __launch_bounds__(256)
void vocab_gemm_kernel(const __bf16* __restrict__ feature_bf, const __bf16* __restrict__ proW,
                       const float* __restrict__ pro_b, float* __restrict__ logits) {
  __shared__ __bf16 btile[2][128 * VK];   // 2 x 16KB
  const int lane = threadIdx.x & 31;
  const int w = threadIdx.x >> 5;
  const int row_base = blockIdx.x * 128;          // first weight row of this block

  v8f acc0 = {0.f, 0.f, 0.f, 0.f, 0.f, 0.f, 0.f, 0.f};
  v8f acc1 = acc0;

  // prime: chunk 0 -> buffer 0
  if (w == 0) {
    tdm_load_tile_bf16((unsigned)(uintptr_t)&btile[0][0],
                       proW + (size_t)row_base * 1024, VK, 128, 1024);
    __builtin_amdgcn_s_wait_tensorcnt(0);
  }
  __syncthreads();

  for (int chunk = 0; chunk < VNC; ++chunk) {
    const int buf = chunk & 1;
    // issue next chunk into the other buffer (freed by last iteration's barrier)
    if (w == 0) {
      if (chunk + 1 < VNC) {
        tdm_load_tile_bf16((unsigned)(uintptr_t)&btile[buf ^ 1][0],
                           proW + (size_t)row_base * 1024 + (chunk + 1) * VK,
                           VK, 128, 1024);
        __builtin_amdgcn_s_wait_tensorcnt(1);   // current chunk's data is resident
      } else {
        __builtin_amdgcn_s_wait_tensorcnt(0);
      }
    }
    __syncthreads();   // btile[buf] visible to all waves

    const int ks0 = chunk * VK;
    v16bf a00 = load_frag(feature_bf, 0,  1024, ks0);
    v16bf a01 = load_frag(feature_bf, 16, 1024, ks0);
    v16bf a10 = load_frag(feature_bf, 0,  1024, ks0 + 32);
    v16bf a11 = load_frag(feature_bf, 16, 1024, ks0 + 32);
    v16bf b0 = load_frag(&btile[buf][0], w * 16, VK, 0);
    v16bf b1 = load_frag(&btile[buf][0], w * 16, VK, 32);
    acc0 = wmma_bf16(a00, b0, acc0);
    acc1 = wmma_bf16(a01, b0, acc1);
    acc0 = wmma_bf16(a10, b1, acc0);
    acc1 = wmma_bf16(a11, b1, acc1);
    __syncthreads();   // all waves done with btile[buf] before it is overwritten
  }

  const int n0 = row_base + w * 16;
  const int col = n0 + (lane & 15);
  const float bias = (col < Vc) ? pro_b[col] : 0.0f;
  const int mbb = (lane >> 4) << 3;
#pragma unroll
  for (int r = 0; r < 8; ++r) {
    logits[(size_t)(mbb + r) * NPAD + col] = acc0[r] + bias;
    logits[(size_t)(16 + mbb + r) * NPAD + col] = acc1[r] + bias;
  }
}

// Per-row: softmax over V, mix with pointer dist, Gumbel-max categorical sample,
// log-prob of sampled token. One block of 1024 threads per batch row.
__global__ __launch_bounds__(1024)
void finalize_kernel(const float* __restrict__ logits, const float* __restrict__ pointer,
                     const float* __restrict__ pgen, int step, int* __restrict__ nxt,
                     float* __restrict__ d_out) {
  __shared__ float red[1024];
  __shared__ float keyS[1024];
  __shared__ float fvS[1024];
  __shared__ int   idS[1024];
  const int b = blockIdx.x, tid = threadIdx.x;
  const float* lrow = logits + (size_t)b * NPAD;
  const float* prow = pointer + (size_t)b * Vc;
  const float pg = pgen[b];

  float lm = -1e30f;
  for (int v = tid; v < Vc; v += 1024) lm = fmaxf(lm, lrow[v]);
  red[tid] = lm; __syncthreads();
  for (int s = 512; s > 0; s >>= 1) { if (tid < s) red[tid] = fmaxf(red[tid], red[tid + s]); __syncthreads(); }
  const float M = red[0]; __syncthreads();

  float ls = 0.f;
  for (int v = tid; v < Vc; v += 1024) ls += __expf(lrow[v] - M);
  red[tid] = ls; __syncthreads();
  for (int s = 512; s > 0; s >>= 1) { if (tid < s) red[tid] += red[tid + s]; __syncthreads(); }
  const float inv = 1.0f / red[0]; __syncthreads();

  float fs = 0.f, bestk = -1e30f, bestf = 0.f;
  int besti = 0;
  for (int v = tid; v < Vc; v += 1024) {
    float fin = pg * __expf(lrow[v] - M) * inv + (1.0f - pg) * prow[v] + EPSC;
    fs += fin;
    unsigned hsh = hash3((unsigned)step, (unsigned)b, (unsigned)v);
    float u = (float)(hsh >> 8) * (1.0f / 16777216.0f) + 1e-7f;
    float g = -__logf(-__logf(u));
    float key = __logf(fin) + g;
    if (key > bestk) { bestk = key; besti = v; bestf = fin; }
  }
  red[tid] = fs; keyS[tid] = bestk; fvS[tid] = bestf; idS[tid] = besti;
  __syncthreads();
  for (int s = 512; s > 0; s >>= 1) {
    if (tid < s) {
      red[tid] += red[tid + s];
      if (keyS[tid + s] > keyS[tid]) {
        keyS[tid] = keyS[tid + s]; fvS[tid] = fvS[tid + s]; idS[tid] = idS[tid + s];
      }
    }
    __syncthreads();
  }
  if (tid == 0) {
    nxt[b] = idS[0];
    d_out[b * MAXLEN + step] = (float)idS[0];                       // ys.T
    d_out[Bc * MAXLEN + b * MAXLEN + step] = __logf(fvS[0]) - __logf(red[0]);  // vals.T
  }
}

// ---------------------------------------------------------------------------
// Host entry
// ---------------------------------------------------------------------------
extern "C" void kernel_launch(void* const* d_in, const int* in_sizes, int n_in,
                              void* d_out_v, int out_size, void* d_ws, size_t ws_size,
                              hipStream_t stream) {
  const int*   src          = (const int*)d_in[0];
  // d_in[1] src_mask: unused by the reference math
  const int*   start_symbol = (const int*)d_in[3];
  const float* emb_W        = (const float*)d_in[4];
  const float* encWihF      = (const float*)d_in[5];
  const float* encWhhF      = (const float*)d_in[6];
  const float* encBihF      = (const float*)d_in[7];
  const float* encBhhF      = (const float*)d_in[8];
  const float* encWihB      = (const float*)d_in[9];
  const float* encWhhB      = (const float*)d_in[10];
  const float* encBihB      = (const float*)d_in[11];
  const float* encBhhB      = (const float*)d_in[12];
  const float* decWih       = (const float*)d_in[13];
  const float* decWhh       = (const float*)d_in[14];
  const float* decBih       = (const float*)d_in[15];
  const float* decBhh       = (const float*)d_in[16];
  const float* proW         = (const float*)d_in[17];
  const float* proB         = (const float*)d_in[18];
  const float* pgenW        = (const float*)d_in[19];
  const float* pgenB        = (const float*)d_in[20];
  const float* covW         = (const float*)d_in[21];
  float* out = (float*)d_out_v;

  char* ws = (char*)d_ws;
  size_t off = 0;
  auto alloc = [&](size_t bytes) -> char* {
    off = (off + 255) & ~(size_t)255;
    char* p = ws + off;
    off += bytes;
    return p;
  };

  __bf16* proW_bf    = (__bf16*)alloc((size_t)NPAD * 1024 * 2);
  __bf16* eWihF      = (__bf16*)alloc((size_t)1024 * 128 * 2);
  __bf16* eWhhF      = (__bf16*)alloc((size_t)1024 * 256 * 2);
  __bf16* eWihB      = (__bf16*)alloc((size_t)1024 * 128 * 2);
  __bf16* eWhhB      = (__bf16*)alloc((size_t)1024 * 256 * 2);
  __bf16* dWih_bf    = (__bf16*)alloc((size_t)2048 * 128 * 2);
  __bf16* dWhh_bf    = (__bf16*)alloc((size_t)2048 * 512 * 2);
  __bf16* xs_bf      = (__bf16*)alloc((size_t)Lc * Bc * Ec * 2);
  float*  memoryB    = (float*)alloc((size_t)Bc * Lc * 512 * 4);
  float*  dec_h      = (float*)alloc((size_t)Bc * 512 * 4);
  float*  dec_c      = (float*)alloc((size_t)Bc * 512 * 4);
  __bf16* h_bf       = (__bf16*)alloc((size_t)Bc * 512 * 2);
  float*  gates      = (float*)alloc((size_t)Bc * 2048 * 4);
  float*  emb_f      = (float*)alloc((size_t)Bc * 128 * 4);
  __bf16* emb_bf     = (__bf16*)alloc((size_t)Bc * 128 * 2);
  __bf16* feature_bf = (__bf16*)alloc((size_t)Bc * 1024 * 2);
  float*  ctx        = (float*)alloc((size_t)Bc * 512 * 4);
  float*  att        = (float*)alloc((size_t)Bc * Lc * 4);
  float*  cov        = (float*)alloc((size_t)Bc * Lc * 4);
  float*  pgenbuf    = (float*)alloc((size_t)Bc * 4);
  float*  logits     = (float*)alloc((size_t)Bc * NPAD * 4);
  float*  pointer    = (float*)alloc((size_t)Bc * Vc * 4);
  int*    nxt        = (int*)alloc((size_t)Bc * 4);
  float*  covloss    = (float*)alloc(4);
  (void)ws_size; (void)out_size; (void)n_in; (void)in_sizes;

  const int T = 256;
  auto cdiv = [](long n, int bsz) { return (int)((n + bsz - 1) / bsz); };

  // Weight conversions (bf16): halves dominant traffic, makes pro_W L2-resident.
  convert_bf16_kernel<<<cdiv(1024 * 128, T), T, 0, stream>>>(encWihF, eWihF, 1024 * 128);
  convert_bf16_kernel<<<cdiv(1024 * 256, T), T, 0, stream>>>(encWhhF, eWhhF, 1024 * 256);
  convert_bf16_kernel<<<cdiv(1024 * 128, T), T, 0, stream>>>(encWihB, eWihB, 1024 * 128);
  convert_bf16_kernel<<<cdiv(1024 * 256, T), T, 0, stream>>>(encWhhB, eWhhB, 1024 * 256);
  convert_bf16_kernel<<<cdiv(2048 * 128, T), T, 0, stream>>>(decWih, dWih_bf, 2048 * 128);
  convert_bf16_kernel<<<cdiv(2048 * 512, T), T, 0, stream>>>(decWhh, dWhh_bf, 2048 * 512);
  convert_proW_kernel<<<4096, T, 0, stream>>>(proW, proW_bf);
  embed_src_kernel<<<cdiv((long)Lc * Bc * Ec, T), T, 0, stream>>>(src, emb_W, xs_bf);

  // Persistent bidirectional encoder (1 block per direction).
  encoder_kernel<<<2, 1024, 0, stream>>>(xs_bf, eWihF, eWhhF, encBihF, encBhhF,
                                         eWihB, eWhhB, encBihB, encBhhB,
                                         memoryB, dec_h, dec_c);

  dec_init_kernel<<<cdiv(Bc * Lc, T), T, 0, stream>>>(start_symbol, nxt, cov, covloss);
  hbf_init_kernel<<<cdiv(Bc * 512, T), T, 0, stream>>>(dec_h, h_bf);

  for (int step = 0; step < MAXLEN; ++step) {
    dec_embed_kernel<<<cdiv(Bc * Ec, T), T, 0, stream>>>(emb_W, nxt, emb_f, emb_bf);
    dec_gates_kernel<<<32, 256, 0, stream>>>(emb_bf, h_bf, dWih_bf, dWhh_bf,
                                             decBih, decBhh, gates);
    dec_cell_kernel<<<cdiv(Bc * 512, T), T, 0, stream>>>(gates, dec_h, dec_c, h_bf, feature_bf);
    attention_kernel<<<Bc, 256, 0, stream>>>(dec_h, memoryB, cov, covW, pgenW, pgenB,
                                             emb_f, att, ctx, feature_bf, pgenbuf, covloss);
    zero_kernel<<<1024, T, 0, stream>>>(pointer, Bc * Vc);
    pointer_scatter_kernel<<<cdiv(Bc * Lc, T), T, 0, stream>>>(src, att, pointer);
    vocab_gemm_kernel<<<NPAD / 128, 256, 0, stream>>>(feature_bf, proW_bf, proB, logits);
    finalize_kernel<<<Bc, 1024, 0, stream>>>(logits, pointer, pgenbuf, step, nxt, out);
  }
  copy1_kernel<<<1, 1, 0, stream>>>(covloss, out + 2 * Bc * MAXLEN);
}
